// GraphConvLayer_48558900248840
// MI455X (gfx1250) — matrix operand order repaired and checked by reading, not compiled, not run on hardware
//
#include <hip/hip_runtime.h>
#include <stdint.h>

// ---------------------------------------------------------------------------
// Chebyshev GraphConv on MI455X (gfx1250, wave32, v_wmma_f32_16x16x32_f16).
// Per (b,t):  Y1 = G@X ; Y2 = 2*(G@Y1) - X ; Out = X@W0 + Y1@W1 + Y2@W2 + bias + X
// All WMMA fragment loads are contiguous 16B/32B vector loads; layout
// transposes happen through per-wave LDS tiles (ds_store_b16 scatter +
// b128 LDS reads), never through global scalar gathers.
// ---------------------------------------------------------------------------

typedef __attribute__((ext_vector_type(16))) _Float16 v16h;
typedef __attribute__((ext_vector_type(8)))  _Float16 v8h;
typedef __attribute__((ext_vector_type(8)))  float    v8f;

#define BB 32
#define CC 64
#define TT 12
#define NNODE 2048
#define KS 3
#define PH 72   // f16 LDS transpose row pitch (halfs): 144B, 16B-aligned rows
#define PF 20   // f32 LDS transpose row pitch (floats): 80B, 16B-aligned rows

// A-tile: row-major source, element (m,k) at base + m*lda + k.
// Contiguous k order (k = 16*hi + s): two adjacent 16B loads per lane.
__device__ __forceinline__ v16h load_a_rm(const _Float16* __restrict__ base,
                                          size_t lda, int m0, int k0,
                                          int lane16, int hi) {
    const _Float16* p = base + (size_t)(m0 + lane16) * lda + (size_t)(k0 + hi * 16);
    union { v16h v; v8h h[2]; } u;
    u.h[0] = *(const v8h*)p;
    u.h[1] = *(const v8h*)(p + 8);
    return u.v;
}

// B-tile: column-contiguous source, element (k,n) at base + n*ldc + k.
// One 32B load per lane.
__device__ __forceinline__ v16h load_b_cc(const _Float16* __restrict__ base,
                                          size_t ldc, int k0, int n0,
                                          int lane16, int hi) {
    const _Float16* p = base + (size_t)(n0 + lane16) * ldc + (size_t)(k0 + hi * 16);
    return *(const v16h*)(p);
}

// ---------------------------------------------------------------------------
// xprep: read x (f32) once; write f16 in natural [b][c][t][n] layout (Xcn,
// stage-1 B source) and [bt][n][c] layout (Xnc, stage-3 B source) via a
// per-wave LDS transpose.  Grid (16, B*T), block 256 (8 waves x 16 nodes).
// ---------------------------------------------------------------------------
__global__ __launch_bounds__(256)
void xprep(const float* __restrict__ xf, _Float16* __restrict__ Xcn,
           _Float16* __restrict__ Xnc) {
    const int lane = threadIdx.x & 31;
    const int wave = threadIdx.x >> 5;
    const int bt   = blockIdx.y;
    const int b    = bt / TT;
    const int t    = bt - b * TT;
    const int n0   = blockIdx.x * 128 + wave * 16;

    __shared__ _Float16 ltr[8][16 * PH];
    _Float16* wls = ltr[wave];

#pragma unroll
    for (int hhalf = 0; hhalf < 2; ++hhalf) {
        const int c = lane + hhalf * 32;
        const size_t off = (((size_t)b * CC + c) * TT + t) * NNODE + n0;
        union { v16h v; _Float16 e[16]; } hv;
#pragma unroll
        for (int q = 0; q < 4; ++q) {
            float4 f = *(const float4*)(xf + off + q * 4);
            hv.e[q * 4 + 0] = (_Float16)f.x;
            hv.e[q * 4 + 1] = (_Float16)f.y;
            hv.e[q * 4 + 2] = (_Float16)f.z;
            hv.e[q * 4 + 3] = (_Float16)f.w;
        }
        *(v16h*)(Xcn + off) = hv.v;          // natural layout, 32B store
#pragma unroll
        for (int i = 0; i < 16; ++i) wls[i * PH + c] = hv.e[i];
    }

    asm volatile("s_wait_dscnt 0x0" ::: "memory");  // per-wave tile: DScnt only

    _Float16* dst = Xnc + (size_t)bt * NNODE * CC;
#pragma unroll
    for (int i = 0; i < 4; ++i) {
        const int idx = i * 32 + lane;      // 0..127
        const int row = idx >> 3;           // 0..15
        const int ch  = (idx & 7) * 8;      // 0..56
        v8h v = *(const v8h*)(wls + row * PH + ch);
        *(v8h*)(dst + (size_t)(n0 + row) * CC + ch) = v;  // coalesced 16B
    }
}

// ---------------------------------------------------------------------------
// Stage 1: Y1 = G@X   -> Y1cn [bt][c][n] (direct) + Y1nc [bt][n][c] (LDS tr.)
// Stage 2: Y2 = 2*(G@Y1) - X -> Y2nc [bt][n][c] only.
// Grid (16, B*T), block 256; each wave owns 16 node-rows x all 64 channels.
// ---------------------------------------------------------------------------
template <int STAGE>
__global__ __launch_bounds__(256)
void cheb_gso_gemm(const _Float16* __restrict__ Gh,
                   const _Float16* __restrict__ Bsrc,  // Xcn (1) / Y1cn (2)
                   const float*    __restrict__ xf,    // residual (stage 2)
                   _Float16*       __restrict__ Ycn,   // stage 1 only
                   _Float16*       __restrict__ Ync) {
    const int lane   = threadIdx.x & 31;
    const int lane16 = lane & 15;
    const int hi     = lane >> 4;
    const int wave   = threadIdx.x >> 5;
    const int bt     = blockIdx.y;
    const int b      = bt / TT;
    const int t      = bt - b * TT;
    const int m0     = blockIdx.x * 128 + wave * 16;

    const _Float16* bbase;
    size_t ldb;
    if (STAGE == 1) {
        bbase = Bsrc + (size_t)b * CC * TT * NNODE + (size_t)t * NNODE;
        ldb   = (size_t)TT * NNODE;
    } else {
        bbase = Bsrc + (size_t)bt * CC * NNODE;
        ldb   = NNODE;
    }

    v8f acc[4] = {};
    for (int k0 = 0; k0 < NNODE; k0 += 32) {
        v16h a = load_a_rm(Gh, NNODE, m0, k0, lane16, hi);
        if (k0 + 256 < NNODE)
            __builtin_prefetch(Gh + (size_t)(m0 + lane16) * NNODE + k0 + 256, 0, 1);
#pragma unroll
        for (int ct = 0; ct < 4; ++ct) {
            v16h bm = load_b_cc(bbase, ldb, k0, ct * 16, lane16, hi);
            acc[ct] = __builtin_amdgcn_wmma_f32_16x16x32_f16(
                false, a, false, bm, (short)0, acc[ct], false, false);
        }
    }

    __shared__ _Float16 ltr[8][16 * PH];
    _Float16* wls = ltr[wave];
    const int h0 = m0 + hi * 8;

#pragma unroll
    for (int ct = 0; ct < 4; ++ct) {
        const int c = ct * 16 + lane16;
        float vals[8];
        if (STAGE == 2) {
            const float* xp = xf + (((size_t)b * CC + c) * TT + t) * NNODE + h0;
            float4 xa = *(const float4*)xp;
            float4 xb = *(const float4*)(xp + 4);
            vals[0] = 2.f * acc[ct][0] - xa.x; vals[1] = 2.f * acc[ct][1] - xa.y;
            vals[2] = 2.f * acc[ct][2] - xa.z; vals[3] = 2.f * acc[ct][3] - xa.w;
            vals[4] = 2.f * acc[ct][4] - xb.x; vals[5] = 2.f * acc[ct][5] - xb.y;
            vals[6] = 2.f * acc[ct][6] - xb.z; vals[7] = 2.f * acc[ct][7] - xb.w;
        } else {
#pragma unroll
            for (int r = 0; r < 8; ++r) vals[r] = acc[ct][r];
        }
        if (STAGE == 1) {  // direct [c][n] store: 8 consecutive nodes -> 16B
            union { v8h v; _Float16 e[8]; } o;
#pragma unroll
            for (int r = 0; r < 8; ++r) o.e[r] = (_Float16)vals[r];
            *(v8h*)(Ycn + (size_t)bt * CC * NNODE + (size_t)c * NNODE + h0) = o.v;
        }
#pragma unroll
        for (int r = 0; r < 8; ++r)  // LDS scatter as [n_local][c]
            wls[(hi * 8 + r) * PH + c] = (_Float16)vals[r];
    }

    asm volatile("s_wait_dscnt 0x0" ::: "memory");

    _Float16* dst = Ync + (size_t)bt * NNODE * CC;
#pragma unroll
    for (int i = 0; i < 4; ++i) {
        const int idx = i * 32 + lane;
        const int row = idx >> 3;
        const int ch  = (idx & 7) * 8;
        v8h v = *(const v8h*)(wls + row * PH + ch);
        *(v8h*)(dst + (size_t)(m0 + row) * CC + ch) = v;
    }
}

// ---------------------------------------------------------------------------
// Stage 3: Out^T[j,n] = sum_k W_k^T @ Y_k^T, then +bias +x residual.
// A = pre-transposed W ([k][j][c], contiguous), B = Yk in [n][c] (contiguous).
// D tiles transposed through per-wave f32 LDS so output stores are float4.
// ---------------------------------------------------------------------------
__global__ __launch_bounds__(256)
void cheb_out_gemm(const _Float16* __restrict__ Xnc,
                   const _Float16* __restrict__ Y1nc,
                   const _Float16* __restrict__ Y2nc,
                   const _Float16* __restrict__ WhT,
                   const float*    __restrict__ bias,
                   const float*    __restrict__ xf,
                   float*          __restrict__ out) {
    const int lane   = threadIdx.x & 31;
    const int lane16 = lane & 15;
    const int hi     = lane >> 4;
    const int wave   = threadIdx.x >> 5;
    const int bt     = blockIdx.y;
    const int b      = bt / TT;
    const int t      = bt - b * TT;
    const int n0     = blockIdx.x * 128 + wave * 16;  // node-column tile

    v8f acc[4] = {};  // ct = j-tile (4 x 16 output channels)
#pragma unroll
    for (int ks = 0; ks < KS; ++ks) {
        const _Float16* asrc = WhT + (size_t)ks * CC * CC;
        const _Float16* bsrc =
            (ks == 0 ? Xnc : (ks == 1 ? Y1nc : Y2nc)) + (size_t)bt * NNODE * CC;
#pragma unroll
        for (int k0 = 0; k0 < CC; k0 += 32) {
            v16h bm = load_b_cc(bsrc, CC, k0, n0, lane16, hi);  // shared by 4 WMMAs
#pragma unroll
            for (int ct = 0; ct < 4; ++ct) {
                v16h a = load_a_rm(asrc, CC, ct * 16, k0, lane16, hi);
                acc[ct] = __builtin_amdgcn_wmma_f32_16x16x32_f16(
                    false, a, false, bm, (short)0, acc[ct], false, false);
            }
        }
    }

    __shared__ float ftr[8][64 * PF];
    float* fls = ftr[wave];
#pragma unroll
    for (int ct = 0; ct < 4; ++ct)
#pragma unroll
        for (int r = 0; r < 8; ++r)
            fls[(ct * 16 + hi * 8 + r) * PF + lane16] = acc[ct][r];

    asm volatile("s_wait_dscnt 0x0" ::: "memory");

#pragma unroll
    for (int i = 0; i < 8; ++i) {
        const int idx = i * 32 + lane;   // 0..255
        const int j   = idx >> 2;        // 0..63
        const int q   = (idx & 3) * 4;   // 0,4,8,12
        float4 v = *(const float4*)(fls + j * PF + q);
        const size_t off = (((size_t)b * CC + j) * TT + t) * NNODE + n0 + q;
        float4 xr = *(const float4*)(xf + off);
        const float bj = bias[j];
        float4 o;
        o.x = v.x + bj + xr.x; o.y = v.y + bj + xr.y;
        o.z = v.z + bj + xr.z; o.w = v.w + bj + xr.w;
        *(float4*)(out + off) = o;
    }
}

// ---------------------------------------------------------------------------
// Small converts
// ---------------------------------------------------------------------------
__global__ void cvt_f32_f16(const float* __restrict__ src,
                            _Float16* __restrict__ dst, size_t n) {
    size_t i = (size_t)blockIdx.x * blockDim.x + threadIdx.x;
    const size_t stride = (size_t)gridDim.x * blockDim.x;
    for (; i < n; i += stride) dst[i] = (_Float16)src[i];
}

__global__ void wprep(const float* __restrict__ w, _Float16* __restrict__ WhT) {
    const int i = blockIdx.x * 256 + threadIdx.x;  // out index [k][j][c]
    if (i < KS * CC * CC) {
        const int k = i / (CC * CC);
        const int r = i - k * CC * CC;
        const int j = r >> 6;
        const int c = r & 63;
        WhT[i] = (_Float16)w[k * CC * CC + c * CC + j];
    }
}

extern "C" void kernel_launch(void* const* d_in, const int* in_sizes, int n_in,
                              void* d_out, int out_size, void* d_ws, size_t ws_size,
                              hipStream_t stream) {
    const float* x    = (const float*)d_in[0];
    const float* gso  = (const float*)d_in[1];
    const float* w    = (const float*)d_in[2];
    const float* bias = (const float*)d_in[3];
    float* out = (float*)d_out;

    const size_t SX = (size_t)BB * CC * TT * NNODE;  // 50,331,648 elems
    const size_t SG = (size_t)NNODE * NNODE;
    const size_t SW = (size_t)KS * CC * CC;

    uintptr_t p = ((uintptr_t)d_ws + 255) & ~(uintptr_t)255;
    _Float16* Xcn  = (_Float16*)p; p = (p + SX * 2 + 255) & ~(uintptr_t)255;
    _Float16* Gh   = (_Float16*)p; p = (p + SG * 2 + 255) & ~(uintptr_t)255;
    _Float16* WhT  = (_Float16*)p; p = (p + SW * 2 + 255) & ~(uintptr_t)255;
    _Float16* Y1cn = (_Float16*)p; p = (p + SX * 2 + 255) & ~(uintptr_t)255;
    _Float16* Xnc  = (_Float16*)p; p = (p + SX * 2 + 255) & ~(uintptr_t)255;
    _Float16* Y1nc = (_Float16*)p;
    _Float16* Y2nc = Xcn;  // Xcn is dead after stage 1 -> reuse for Y2

    dim3 grid(NNODE / 128, BB * TT);  // 16 x 384

    cvt_f32_f16<<<2048, 256, 0, stream>>>(gso, Gh, SG);
    wprep<<<48, 256, 0, stream>>>(w, WhT);
    xprep<<<grid, 256, 0, stream>>>(x, Xcn, Xnc);

    cheb_gso_gemm<1><<<grid, 256, 0, stream>>>(Gh, Xcn, nullptr, Y1cn, Y1nc);
    cheb_gso_gemm<2><<<grid, 256, 0, stream>>>(Gh, Y1cn, x, nullptr, Y2nc);
    cheb_out_gemm<<<grid, 256, 0, stream>>>(Xnc, Y1nc, Y2nc, WhT, bias, x, out);

    (void)in_sizes; (void)n_in; (void)out_size; (void)ws_size;
}
